// EGCL_7799660609601
// MI455X (gfx1250) — matrix operand order repaired and checked by reading, not compiled, not run on hardware
//
#include <hip/hip_runtime.h>
#include <hip/hip_bf16.h>

typedef _Float16 half_t;
typedef __attribute__((ext_vector_type(16))) _Float16 v16h;
typedef __attribute__((ext_vector_type(8)))  _Float16 v8h;
typedef __attribute__((ext_vector_type(8)))  float    v8f;

constexpr int BB   = 4;
constexpr int NN   = 512;
constexpr int FDIM = 128;
constexpr int HIDD = 256;
constexpr int MDIM = 128;
constexpr int WAVES = 8;   // waves per workgroup (wave32)
constexpr int JC    = 128; // j-chunk of c staged in LDS

// silu with fast hardware reciprocal (feeds f16 WMMA inputs; v_rcp_f32 has
// ~1ulp accuracy which is far below f16 quantization error).
__device__ __forceinline__ float siluf(float v) {
    return v * __builtin_amdgcn_rcpf(1.0f + __expf(-v));
}

__device__ __forceinline__ v8f wmma_f16(v16h a, v16h b, v8f c) {
    return __builtin_amdgcn_wmma_f32_16x16x32_f16(false, a, false, b,
                                                  (short)0, c, false, false);
}

// CDNA5 async memory->LDS copy (16B per lane), tracked by ASYNCcnt.
// LDS address operand is the low 32 bits of the generic (flat) pointer.
__device__ __forceinline__ void async_copy16(void* lds_dst, const void* gsrc) {
    uint32_t la = (uint32_t)(uintptr_t)lds_dst;
    asm volatile("global_load_async_to_lds_b128 %0, %1, off"
                 :: "v"(la), "v"(gsrc) : "memory");
}
__device__ __forceinline__ void async_wait0() {
    asm volatile("s_wait_asynccnt 0x0" ::: "memory");
}

// A-fragment (16x32 f16) from a row-major f16 tile with row stride `ld`
// halves, K origin kb. Per CDNA5 ISA 7.12.2: lane L holds row M=L&15;
// its 16 halves are K = [kg, kg+8) and [16+kg, 16+kg+8), kg = (L>>4)*8,
// i.e. two contiguous 16-byte runs.
__device__ __forceinline__ v16h load_afrag(const half_t* src, int ld, int kb) {
    int lane = threadIdx.x & 31;
    int m    = lane & 15;
    int kg   = (lane >> 4) << 3;
    const half_t* p = src + m * ld + kb + kg;
    v8h lo = *(const v8h*)(p);
    v8h hi = *(const v8h*)(p + 16);
    v16h r;
#pragma unroll
    for (int t = 0; t < 8; ++t) { r[t] = lo[t]; r[8 + t] = hi[t]; }
    return r;
}

// ---------------------------------------------------------------------------
// Kernel 1: per-node precompute.
//   a[row][t] = h[row] . We1[0:128, t] + be1[t]        (f32)
//   c[row][t] = h[row] . We1[128:256, t]               (f16)
//   concat[row][0:128] = f16(h[row])
// ---------------------------------------------------------------------------
__global__ __launch_bounds__(256)
void egcl_pre_kernel(const float* __restrict__ h,
                     const float* __restrict__ We1,
                     const float* __restrict__ be1,
                     float* __restrict__ a_ws,
                     half_t* __restrict__ c_ws,
                     half_t* __restrict__ concat_ws) {
    __shared__ float hrow[FDIM];
    int row = blockIdx.x;              // b*NN + i
    int tid = threadIdx.x;             // 256 == HIDD
    if (tid < FDIM) hrow[tid] = h[row * FDIM + tid];
    __syncthreads();
    float aa = be1[tid];
    float cc = 0.0f;
#pragma unroll 4
    for (int k = 0; k < FDIM; ++k) {
        float hv = hrow[k];
        aa += hv * We1[k * HIDD + tid];
        cc += hv * We1[(FDIM + k) * HIDD + tid];
    }
    a_ws[row * HIDD + tid] = aa;
    c_ws[row * HIDD + tid] = (half_t)cc;
    if (tid < FDIM) concat_ws[row * (FDIM + MDIM) + tid] = (half_t)hrow[tid];
}

// ---------------------------------------------------------------------------
// Kernel 2: fused edge pipeline. One wave per node i, 16-edge j tiles.
// ---------------------------------------------------------------------------
constexpr size_t EDGE_SMEM =
    (size_t)64 * 512 * 2 +      // We2s  (frag-major swizzled, 8kk x 8nt)
    (size_t)64 * 512 * 2 +      // Wx1s  (frag-major swizzled, 4kk x 16nt)
    (size_t)JC * HIDD * 2 +     // cs    (c chunk, f16)
    (size_t)WAVES * HIDD * 4 +  // as_   (a rows for this WG, f32)
    (size_t)HIDD * 4 +          // wrs
    (size_t)MDIM * 4 +          // be2s
    (size_t)HIDD * 4 +          // bx1s
    (size_t)NN * 3 * 4 +        // xs
    (size_t)NN * 4 +            // ms (mask as float)
    (size_t)HIDD * 2 +          // wx2s (f16)
    (size_t)WAVES * 16 * 128 * 2 + // sm   (m tile restage per wave, f16)
    (size_t)WAVES * 32 * 8 * 4 +   // red  (per-wave reduction scratch)
    (size_t)WAVES * 32 * 4;        // r2adj

__global__ __launch_bounds__(256)
void egcl_edge_kernel(const float* __restrict__ x,
                      const unsigned char* __restrict__ mask,
                      const float* __restrict__ We1,
                      const float* __restrict__ We2,
                      const float* __restrict__ be2,
                      const float* __restrict__ Wx1,
                      const float* __restrict__ bx1,
                      const float* __restrict__ Wx2,
                      const float* __restrict__ bx2,
                      const float* __restrict__ a_ws,
                      const half_t* __restrict__ c_ws,
                      half_t* __restrict__ concat_ws,
                      float* __restrict__ x_out) {
    extern __shared__ char smem[];
    char* ptr = smem;
    half_t* We2s = (half_t*)ptr; ptr += 64 * 512 * 2;
    half_t* Wx1s = (half_t*)ptr; ptr += 64 * 512 * 2;
    half_t* cs   = (half_t*)ptr; ptr += JC * HIDD * 2;
    float*  as_  = (float*)ptr;  ptr += WAVES * HIDD * 4;
    float*  wrs  = (float*)ptr;  ptr += HIDD * 4;
    float*  be2s = (float*)ptr;  ptr += MDIM * 4;
    float*  bx1s = (float*)ptr;  ptr += HIDD * 4;
    float*  xs   = (float*)ptr;  ptr += NN * 3 * 4;
    float*  ms   = (float*)ptr;  ptr += NN * 4;
    half_t* wx2s = (half_t*)ptr; ptr += HIDD * 2;
    half_t* sm   = (half_t*)ptr; ptr += WAVES * 16 * 128 * 2;
    float*  red  = (float*)ptr;  ptr += WAVES * 32 * 8 * 4;
    float*  r2adj = (float*)ptr;

    int tid  = threadIdx.x;
    int wave = tid >> 5;
    int lane = tid & 31;
    int b    = blockIdx.x >> 6;              // NN/WAVES = 64 blocks per batch
    int i0   = (blockIdx.x & 63) * WAVES;
    int i    = i0 + wave;

    // ---- stage weights (frag-major swizzle: lane's 16 halves contiguous) ----
    for (int s = tid; s < 64 * 512; s += 256) {
        int frag = s >> 9, rem = s & 511, l = rem >> 4, hh = rem & 15;
        int kk = frag >> 3, nt = frag & 7;                 // 8kk x 8nt
        int k = kk * 32 + ((l >> 4) << 3) + (hh & 7) + ((hh >> 3) << 4);
        int n = nt * 16 + (l & 15);
        We2s[s] = (half_t)We2[k * MDIM + n];               // We2: 256x128
    }
    for (int s = tid; s < 64 * 512; s += 256) {
        int frag = s >> 9, rem = s & 511, l = rem >> 4, hh = rem & 15;
        int kk = frag >> 4, nt = frag & 15;                // 4kk x 16nt
        int k = kk * 32 + ((l >> 4) << 3) + (hh & 7) + ((hh >> 3) << 4); // k < 128
        int n = nt * 16 + (l & 15);
        Wx1s[s] = (half_t)Wx1[k * HIDD + n];               // Wx1: 128x256
    }
    for (int s = tid; s < HIDD; s += 256) {
        wrs[s]  = We1[2 * FDIM * HIDD + s];                // r2 row of We1
        bx1s[s] = bx1[s];
        wx2s[s] = (half_t)Wx2[s];
    }
    if (tid < MDIM) be2s[tid] = be2[tid];
    for (int s = tid; s < NN * 3; s += 256) xs[s] = x[b * NN * 3 + s];
    for (int s = tid; s < NN; s += 256) ms[s] = mask[b * NN + s] ? 1.0f : 0.0f;
    {
        const float4* src = (const float4*)(a_ws + (size_t)(b * NN + i0) * HIDD);
        float4* dst = (float4*)as_;
        for (int s = tid; s < WAVES * HIDD / 4; s += 256) dst[s] = src[s];
    }
    __syncthreads();

    float xi0 = xs[i * 3 + 0], xi1 = xs[i * 3 + 1], xi2 = xs[i * 3 + 2];
    float msi = ms[i];
    float bx2v = bx2[0];

    float mi[8];
#pragma unroll
    for (int t = 0; t < 8; ++t) mi[t] = 0.0f;
    float xacc0 = 0.0f, xacc1 = 0.0f, xacc2 = 0.0f, degacc = 0.0f;

    float*  red_w = red + wave * 32 * 8;
    float*  ra    = r2adj + wave * 32;
    half_t* sm_w  = sm + wave * 16 * 128;
    const float* arow = as_ + wave * HIDD;

    int mloc = lane & 15;
    int kg   = (lane >> 4) << 3;
    int hi8  = (lane >> 4) << 3;
    int col0 = lane & 15;
    v8f zero8 = {0.f, 0.f, 0.f, 0.f, 0.f, 0.f, 0.f, 0.f};

    for (int jc0 = 0; jc0 < NN; jc0 += JC) {
        __syncthreads();  // protect cs reuse
        {
            // CDNA5 async global->LDS staging of the c chunk (no VGPR bounce).
            const char* src = (const char*)(c_ws + (size_t)(b * NN + jc0) * HIDD);
            char* dst = (char*)cs;
            for (int s = tid * 16; s < JC * HIDD * 2; s += 256 * 16)
                async_copy16(dst + s, src + s);
            async_wait0();
        }
        __syncthreads();

        for (int jt = 0; jt < JC; jt += 16) {
            // geometry for this wave's 16 edges (i, jb..jb+15)
            int j = jc0 + jt + mloc;
            float dx = xi0 - xs[j * 3 + 0];
            float dy = xi1 - xs[j * 3 + 1];
            float dz = xi2 - xs[j * 3 + 2];
            float pm  = msi * ms[j];
            float r2v = (dx * dx + dy * dy + dz * dz) * pm;
            float adjf = (j == i) ? 0.0f : pm;
            if (lane < 16) { ra[mloc] = r2v; ra[16 + mloc] = adjf; }
            __syncthreads();

            // ---- layer 2 GEMM: m = silu(m1 @ We2 + be2), m1 built on the fly
            const half_t* crow = cs + (jt + mloc) * HIDD;
            v8f acc2[8];
#pragma unroll
            for (int t = 0; t < 8; ++t) acc2[t] = zero8;
#pragma unroll
            for (int kk = 0; kk < 8; ++kk) {
                int kb0 = kk * 32 + kg, kb1 = kb0 + 16;
                v8h c0 = *(const v8h*)(crow + kb0);
                v8h c1 = *(const v8h*)(crow + kb1);
                v8f a0 = *(const v8f*)(arow + kb0);
                v8f a1 = *(const v8f*)(arow + kb1);
                v8f w0 = *(const v8f*)(wrs + kb0);
                v8f w1 = *(const v8f*)(wrs + kb1);
                v16h A;
#pragma unroll
                for (int t = 0; t < 8; ++t) {
                    A[t]     = (half_t)siluf(a0[t] + (float)c0[t] + r2v * w0[t]);
                    A[8 + t] = (half_t)siluf(a1[t] + (float)c1[t] + r2v * w1[t]);
                }
#pragma unroll
                for (int nt = 0; nt < 8; ++nt) {
                    v16h Bf = *(const v16h*)(We2s + (kk * 8 + nt) * 512 + lane * 16);
                    acc2[nt] = wmma_f16(A, Bf, acc2[nt]);
                }
            }
            // epilogue: silu + adj mask, accumulate m_i, restage m as f16 tile
            float adjr[8];
#pragma unroll
            for (int r = 0; r < 8; ++r) adjr[r] = ra[16 + r + hi8];
#pragma unroll
            for (int nt = 0; nt < 8; ++nt) {
                int col = nt * 16 + col0;
                float bc = be2s[col];
#pragma unroll
                for (int r = 0; r < 8; ++r) {
                    float v = siluf(acc2[nt][r] + bc) * adjr[r];
                    mi[nt] += v;
                    sm_w[(r + hi8) * 128 + col] = (half_t)v;
                }
            }
            __syncthreads();

            // ---- layer x1 GEMM: s = silu(m_ij @ Wx1 + bx1), then dot Wx2
            v16h A3[4];
#pragma unroll
            for (int kk = 0; kk < 4; ++kk) A3[kk] = load_afrag(sm_w, 128, kk * 32);
            float p[8];
#pragma unroll
            for (int r = 0; r < 8; ++r) p[r] = 0.0f;
#pragma unroll
            for (int nt = 0; nt < 16; ++nt) {
                v8f acc3 = zero8;
#pragma unroll
                for (int kk = 0; kk < 4; ++kk) {
                    v16h Bf = *(const v16h*)(Wx1s + (kk * 16 + nt) * 512 + lane * 16);
                    acc3 = wmma_f16(A3[kk], Bf, acc3);
                }
                int col = nt * 16 + col0;
                float bc = bx1s[col];
                float wv = (float)wx2s[col];
#pragma unroll
                for (int r = 0; r < 8; ++r) p[r] += siluf(acc3[r] + bc) * wv;
            }
#pragma unroll
            for (int r = 0; r < 8; ++r) red_w[lane * 8 + r] = p[r];
            __syncthreads();
            if (lane < 16) {
                int mrow = lane;
                int lbase = (mrow < 8) ? 0 : 16;
                int r = mrow & 7;
                float sij = bx2v;
#pragma unroll
                for (int l = 0; l < 16; ++l) sij += red_w[(lbase + l) * 8 + r];
                float av = ra[16 + mrow];
                float sv = tanhf(sij) * 0.1f * av;
                int jj = jc0 + jt + mrow;
                xacc0 += (xi0 - xs[jj * 3 + 0]) * sv;
                xacc1 += (xi1 - xs[jj * 3 + 1]) * sv;
                xacc2 += (xi2 - xs[jj * 3 + 2]) * sv;
                degacc += av;
            }
            __syncthreads();
        }
    }

    // ---- finalize m_i (combine lane halves) into concat buffer (f16) ----
#pragma unroll
    for (int t = 0; t < 8; ++t) red_w[lane * 8 + t] = mi[t];
    __syncthreads();
    if (lane < 16) {
#pragma unroll
        for (int nt = 0; nt < 8; ++nt) {
            float v = red_w[lane * 8 + nt] + red_w[(lane + 16) * 8 + nt];
            int col = nt * 16 + lane;
            concat_ws[(size_t)(b * NN + i) * (FDIM + MDIM) + FDIM + col] = (half_t)v;
        }
    }
    __syncthreads();
    if (lane < 16) {
        red_w[lane * 8 + 0] = xacc0;
        red_w[lane * 8 + 1] = xacc1;
        red_w[lane * 8 + 2] = xacc2;
        red_w[lane * 8 + 3] = degacc;
    }
    __syncthreads();
    if (lane == 0) {
        float s0 = 0, s1 = 0, s2 = 0, dg = 0;
        for (int l = 0; l < 16; ++l) {
            s0 += red_w[l * 8 + 0];
            s1 += red_w[l * 8 + 1];
            s2 += red_w[l * 8 + 2];
            dg += red_w[l * 8 + 3];
        }
        dg = fmaxf(dg, 1.0f);
        size_t ob = (size_t)(b * NN + i) * 3;
        x_out[ob + 0] = xi0 + s0 / dg;
        x_out[ob + 1] = xi1 + s1 / dg;
        x_out[ob + 2] = xi2 + s2 / dg;
    }
}

// ---------------------------------------------------------------------------
// Kernel 3: node MLP + residual + layernorm.
// ---------------------------------------------------------------------------
constexpr size_t NODE_SMEM =
    (size_t)128 * 512 * 2 +        // Wh1s (8kk x 16nt frag-major)
    (size_t)64 * 512 * 2 +         // Wh2s (8kk x 8nt frag-major)
    (size_t)WAVES * 16 * HIDD * 2 + // t stage (reused as f32 hr buffer)
    (size_t)HIDD * 4 + (size_t)FDIM * 4 + (size_t)FDIM * 4 + (size_t)FDIM * 4;

__global__ __launch_bounds__(256)
void egcl_node_kernel(const float* __restrict__ h,
                      const half_t* __restrict__ concat_ws,
                      const float* __restrict__ Wh1,
                      const float* __restrict__ bh1,
                      const float* __restrict__ Wh2,
                      const float* __restrict__ bh2,
                      const float* __restrict__ gamma,
                      const float* __restrict__ beta,
                      float* __restrict__ h_out) {
    extern __shared__ char smem[];
    char* ptr = smem;
    half_t* Wh1s = (half_t*)ptr; ptr += 128 * 512 * 2;
    half_t* Wh2s = (half_t*)ptr; ptr += 64 * 512 * 2;
    half_t* tst  = (half_t*)ptr; ptr += WAVES * 16 * HIDD * 2;
    float*  bh1s = (float*)ptr;  ptr += HIDD * 4;
    float*  bh2s = (float*)ptr;  ptr += FDIM * 4;
    float*  gs   = (float*)ptr;  ptr += FDIM * 4;
    float*  bs   = (float*)ptr;

    int tid = threadIdx.x, wave = tid >> 5, lane = tid & 31;
    for (int s = tid; s < 128 * 512; s += 256) {
        int frag = s >> 9, rem = s & 511, l = rem >> 4, hh = rem & 15;
        int kk = frag >> 4, nt = frag & 15;                // 8kk x 16nt
        int k = kk * 32 + ((l >> 4) << 3) + (hh & 7) + ((hh >> 3) << 4);
        int n = nt * 16 + (l & 15);
        Wh1s[s] = (half_t)Wh1[k * HIDD + n];               // Wh1: 256x256
    }
    for (int s = tid; s < 64 * 512; s += 256) {
        int frag = s >> 9, rem = s & 511, l = rem >> 4, hh = rem & 15;
        int kk = frag >> 3, nt = frag & 7;                 // 8kk x 8nt
        int k = kk * 32 + ((l >> 4) << 3) + (hh & 7) + ((hh >> 3) << 4);
        int n = nt * 16 + (l & 15);
        Wh2s[s] = (half_t)Wh2[k * FDIM + n];               // Wh2: 256x128
    }
    for (int s = tid; s < HIDD; s += 256) bh1s[s] = bh1[s];
    if (tid < FDIM) { bh2s[tid] = bh2[tid]; gs[tid] = gamma[tid]; bs[tid] = beta[tid]; }
    __syncthreads();

    int row0 = blockIdx.x * 128 + wave * 16;
    const half_t* Arow = concat_ws + (size_t)row0 * (FDIM + MDIM);
    half_t* tw = tst + wave * 16 * HIDD;
    int col0 = lane & 15, hi8 = (lane >> 4) << 3;
    v8f zero8 = {0.f, 0.f, 0.f, 0.f, 0.f, 0.f, 0.f, 0.f};

    // layer 1: t = silu([h, m_i] @ Wh1 + bh1)
    v16h A1[8];
#pragma unroll
    for (int kk = 0; kk < 8; ++kk) A1[kk] = load_afrag(Arow, FDIM + MDIM, kk * 32);
#pragma unroll
    for (int nt = 0; nt < 16; ++nt) {
        v8f acc = zero8;
#pragma unroll
        for (int kk = 0; kk < 8; ++kk)
            acc = wmma_f16(A1[kk], *(const v16h*)(Wh1s + (kk * 16 + nt) * 512 + lane * 16), acc);
        int col = nt * 16 + col0;
        float bc = bh1s[col];
#pragma unroll
        for (int r = 0; r < 8; ++r)
            tw[(r + hi8) * HIDD + col] = (half_t)siluf(acc[r] + bc);
    }
    __syncthreads();

    // layer 2: dh = t @ Wh2 + bh2 ; hr = h + dh
    v16h A2[8];
#pragma unroll
    for (int kk = 0; kk < 8; ++kk) A2[kk] = load_afrag(tw, HIDD, kk * 32);
    __syncthreads();
    float* hrbuf = (float*)tw;  // reuse: 16*256 halves == 16*128 floats
#pragma unroll
    for (int nt = 0; nt < 8; ++nt) {
        v8f acc = zero8;
#pragma unroll
        for (int kk = 0; kk < 8; ++kk)
            acc = wmma_f16(A2[kk], *(const v16h*)(Wh2s + (kk * 8 + nt) * 512 + lane * 16), acc);
        int col = nt * 16 + col0;
        float bc = bh2s[col];
#pragma unroll
        for (int r = 0; r < 8; ++r) {
            int row = r + hi8;
            float hr = h[(size_t)(row0 + row) * FDIM + col] + acc[r] + bc;
            hrbuf[row * FDIM + col] = hr;
        }
    }
    __syncthreads();

    // layernorm, one row per lane
    if (lane < 16) {
        int row = lane;
        const float* rp = hrbuf + row * FDIM;
        float mu = 0.0f;
        for (int c = 0; c < FDIM; ++c) mu += rp[c];
        mu *= (1.0f / FDIM);
        float var = 0.0f;
        for (int c = 0; c < FDIM; ++c) { float d = rp[c] - mu; var += d * d; }
        var *= (1.0f / FDIM);
        float inv = rsqrtf(var + 1e-5f);
        float* op = h_out + (size_t)(row0 + row) * FDIM;
        for (int c = 0; c < FDIM; ++c)
            op[c] = (rp[c] - mu) * inv * gs[c] + bs[c];
    }
}

// ---------------------------------------------------------------------------
extern "C" void kernel_launch(void* const* d_in, const int* in_sizes, int n_in,
                              void* d_out, int out_size, void* d_ws, size_t ws_size,
                              hipStream_t stream) {
    (void)in_sizes; (void)n_in; (void)out_size; (void)ws_size;
    const float* h    = (const float*)d_in[0];
    const float* x    = (const float*)d_in[1];
    const unsigned char* mask = (const unsigned char*)d_in[2];
    const float* We1  = (const float*)d_in[3];
    const float* be1  = (const float*)d_in[4];
    const float* We2  = (const float*)d_in[5];
    const float* be2  = (const float*)d_in[6];
    const float* Wx1  = (const float*)d_in[7];
    const float* bx1  = (const float*)d_in[8];
    const float* Wx2  = (const float*)d_in[9];
    const float* bx2  = (const float*)d_in[10];
    const float* Wh1  = (const float*)d_in[11];
    const float* bh1  = (const float*)d_in[12];
    const float* Wh2  = (const float*)d_in[13];
    const float* bh2  = (const float*)d_in[14];
    const float* gamma = (const float*)d_in[15];
    const float* beta  = (const float*)d_in[16];

    size_t rows = (size_t)BB * NN;
    float*  a_ws      = (float*)d_ws;                               // rows*HIDD f32
    half_t* c_ws      = (half_t*)((char*)d_ws + rows * HIDD * 4);   // rows*HIDD f16
    half_t* concat_ws = (half_t*)((char*)d_ws + rows * HIDD * 4 + rows * HIDD * 2);

    float* h_out = (float*)d_out;
    float* x_out = h_out + rows * FDIM;

    egcl_pre_kernel<<<BB * NN, 256, 0, stream>>>(h, We1, be1, a_ws, c_ws, concat_ws);

    egcl_edge_kernel<<<BB * (NN / WAVES), 256, EDGE_SMEM, stream>>>(
        x, mask, We1, We2, be2, Wx1, bx1, Wx2, bx2, a_ws, c_ws, concat_ws, x_out);

    egcl_node_kernel<<<(int)(rows / 128), 256, NODE_SMEM, stream>>>(
        h, concat_ws, Wh1, bh1, Wh2, bh2, gamma, beta, h_out);
}
